// LSTMDQN_23106924052863
// MI455X (gfx1250) — matrix operand order repaired and checked by reading, not compiled, not run on hardware
//
#include <hip/hip_runtime.h>
#include <hip/hip_bf16.h>

typedef __attribute__((ext_vector_type(16))) __bf16 v16bf;
typedef __attribute__((ext_vector_type(8)))  float  v8f;

#define HDIM 64
#define WAVES 4
#define TPB (WAVES * 32)

struct Smem {
  // transposed bf16 weights: layout [K][N]
  __bf16 Wg_t [96 * 256];   // LSTM: rows 0..63 = W_hh^T, 64..68 = W_ih^T, 69..95 = 0
  __bf16 W1_t [64 * 256];
  __bf16 W2_t [256 * 64];
  __bf16 W3_t [64 * 32];
  __bf16 W4_t [32 * 16];
  __bf16 Wh_t [32 * 16];    // head, padded K 16->32, N 3->16
  float  bg[256], b1[256], b2[64], b3[32], b4[16], bhp[16];
  __bf16 actA[WAVES][16 * 96];       // ping buffer (stride 96)
  __bf16 actB[WAVES][16 * 128];      // pong buffer (stride 128)
};

// Branch-free nonlinearities: v_exp_f32 + v_rcp_f32 + bitwise copysign.
__device__ __forceinline__ float sigmoid_fast(float x) {
  return __builtin_amdgcn_rcpf(1.0f + __expf(-x));
}
__device__ __forceinline__ float tanh_fast(float x) {
  const float e = __expf(-2.0f * fabsf(x));          // (0, 1]
  const float t = (1.0f - e) * __builtin_amdgcn_rcpf(1.0f + e);
  return copysignf(t, x);
}

// A-fragment (16x32 bf16 tile) from LDS row-major activations; depends only on
// the k-step, so callers hoist it across all n-tiles.
__device__ __forceinline__ v16bf load_a_frag(const __bf16* act, int aStride, int kb, int lane) {
  const int m = lane & 15;
  const int half8 = (lane >> 4) << 3;  // 0 or 8
  v16bf a;
  unsigned int* ap = reinterpret_cast<unsigned int*>(&a);
#pragma unroll
  for (int v = 0; v < 8; ++v) {
    // 16-bit A layout: lanes 0-15 K pairs {0..7,16..23}; lanes 16-31 {8..15,24..31}
    const int k = kb + ((v & 3) << 1) + half8 + ((v & 4) << 2);
    ap[v] = *reinterpret_cast<const unsigned int*>(act + m * aStride + k);
  }
  return a;
}

// B-fragment load (lane <-> N column, K in A's table order) + one V_WMMA_F32_16X16X32_BF16.
__device__ __forceinline__ v8f wmma_b(v16bf a, const __bf16* Wt, int wStride,
                                      int krowBase, int ntile, int lane, v8f acc) {
  v16bf b;
  const int ncol = (ntile << 4) + (lane & 15);
  const int kh = (lane >> 4) << 3;
#pragma unroll
  for (int s = 0; s < 16; ++s)
    b[s] = Wt[(krowBase + (s & 7) + kh + ((s >> 3) << 4)) * wStride + ncol];
  return __builtin_amdgcn_wmma_f32_16x16x32_bf16(false, a, false, b, (short)0, acc,
                                                 false, false);
}

// GEMM + bias + ReLU over n-tiles [ntStart, ntEnd); output columns rebased to 0.
template <int KSTEPS>
__device__ __forceinline__ void mlp_phase(const __bf16* actIn, int inStride,
                                          __bf16* actOut, int outStride,
                                          const __bf16* Wt, int wStride,
                                          const float* bias, int ntStart, int ntEnd,
                                          int lane) {
  const int m = lane & 15;
  const int half8 = (lane >> 4) << 3;
  v16bf afr[KSTEPS];
#pragma unroll
  for (int ks = 0; ks < KSTEPS; ++ks) afr[ks] = load_a_frag(actIn, inStride, ks * 32, lane);
#pragma clang loop unroll(disable)
  for (int nt = ntStart; nt < ntEnd; ++nt) {
    const float bn = bias[(nt << 4) + m];
    v8f acc;
#pragma unroll
    for (int v = 0; v < 8; ++v) acc[v] = bn;
#pragma unroll
    for (int ks = 0; ks < KSTEPS; ++ks)
      acc = wmma_b(afr[ks], Wt, wStride, ks * 32, nt, lane, acc);
#pragma unroll
    for (int v = 0; v < 8; ++v)
      actOut[(v + half8) * outStride + ((nt - ntStart) << 4) + m] = (__bf16)fmaxf(acc[v], 0.f);
  }
}

// MLP2 partial accumulation: 4 n-tiles (N=64) over one 128-wide K half.
__device__ __forceinline__ void mlp2_accum(const __bf16* actIn /*stride 128*/,
                                           const __bf16* Wt /*[256][64]*/,
                                           int kRowOff, v8f acc2[4], int lane) {
  v16bf afr[4];
#pragma unroll
  for (int ks = 0; ks < 4; ++ks) afr[ks] = load_a_frag(actIn, 128, ks * 32, lane);
#pragma unroll
  for (int nt = 0; nt < 4; ++nt)
#pragma unroll
    for (int ks = 0; ks < 4; ++ks)
      acc2[nt] = wmma_b(afr[ks], Wt, 64, kRowOff + ks * 32, nt, lane, acc2[nt]);
}

__global__ __launch_bounds__(TPB) void lstm_dqn_fused(
    const float* __restrict__ x, const float* __restrict__ h0, const float* __restrict__ c0,
    const float* __restrict__ W_ih, const float* __restrict__ W_hh,
    const float* __restrict__ b_ih, const float* __restrict__ b_hh,
    const float* __restrict__ W1, const float* __restrict__ b1,
    const float* __restrict__ W2, const float* __restrict__ b2,
    const float* __restrict__ W3, const float* __restrict__ b3,
    const float* __restrict__ W4, const float* __restrict__ b4,
    const float* __restrict__ Wh, const float* __restrict__ bh,
    float* __restrict__ out, int B) {
  __shared__ Smem sm;
  const int tid = threadIdx.x;
  const int wave = tid >> 5, lane = tid & 31;
  const int m = lane & 15, half8 = (lane >> 4) << 3;
  const int row0 = (blockIdx.x * WAVES + wave) << 4;

  // ---- prefetch this wave's tile behind the weight-staging preamble ----
#pragma unroll
  for (int p = 0; p < 4; ++p) {
    __builtin_prefetch(h0 + (size_t)row0 * HDIM + p * 1024 + lane * 32, 0, 3);
    __builtin_prefetch(c0 + (size_t)row0 * HDIM + p * 1024 + lane * 32, 0, 3);
  }
  __builtin_prefetch(x + (size_t)row0 * 5 + lane * 8, 0, 3);
  __builtin_prefetch(x + (size_t)B * 5 + (size_t)row0 * 5 + lane * 8, 0, 3);

  // ---- cooperative weight staging: f32 -> bf16, transpose to [K][N] ----
  for (int i = tid; i < 96 * 256; i += TPB) {
    int k = i >> 8, n = i & 255;
    float w = (k < 64) ? W_hh[n * 64 + k] : ((k < 69) ? W_ih[n * 5 + (k - 64)] : 0.f);
    sm.Wg_t[i] = (__bf16)w;
  }
  for (int i = tid; i < 64 * 256; i += TPB) { int k = i >> 8, n = i & 255; sm.W1_t[i] = (__bf16)W1[n * 64 + k]; }
  for (int i = tid; i < 256 * 64; i += TPB) { int k = i >> 6, n = i & 63;  sm.W2_t[i] = (__bf16)W2[n * 256 + k]; }
  for (int i = tid; i < 64 * 32;  i += TPB) { int k = i >> 5, n = i & 31;  sm.W3_t[i] = (__bf16)W3[n * 64 + k]; }
  for (int i = tid; i < 32 * 16;  i += TPB) { int k = i >> 4, n = i & 15;  sm.W4_t[i] = (__bf16)W4[n * 32 + k]; }
  for (int i = tid; i < 32 * 16;  i += TPB) {
    int k = i >> 4, n = i & 15;
    sm.Wh_t[i] = (k < 16 && n < 3) ? (__bf16)Wh[n * 16 + k] : (__bf16)0.f;
  }
  for (int i = tid; i < 256; i += TPB) sm.bg[i] = b_ih[i] + b_hh[i];
  for (int i = tid; i < 256; i += TPB) sm.b1[i] = b1[i];
  for (int i = tid; i < 64;  i += TPB) sm.b2[i] = b2[i];
  for (int i = tid; i < 32;  i += TPB) sm.b3[i] = b3[i];
  for (int i = tid; i < 16;  i += TPB) sm.b4[i] = b4[i];
  for (int i = tid; i < 16;  i += TPB) sm.bhp[i] = (i < 3) ? bh[i] : 0.f;

  __bf16* actA = sm.actA[wave];
  __bf16* actB = sm.actB[wave];

  // ---- per-wave staging ----
  // actA = [ h0 (0..63) | x0 (64..68) | zeros (69..95) ]
  for (int i = lane; i < 16 * 96; i += 32) {
    int r = i / 96, k = i % 96;
    float vv = (k < 64) ? h0[(size_t)(row0 + r) * HDIM + k]
                        : ((k < 69) ? x[(size_t)(row0 + r) * 5 + (k - 64)] : 0.f);
    actA[i] = (__bf16)vv;
  }
  // actB cols 64..95 = [ x1 | zeros ]  (h1 lands in cols 0..63 after step 0)
  for (int i = lane; i < 16 * 32; i += 32) {
    int r = i >> 5, k = i & 31;
    actB[r * 128 + 64 + k] =
        (__bf16)((k < 5) ? x[(size_t)B * 5 + (size_t)(row0 + r) * 5 + k] : 0.f);
  }
  // c0 directly in accumulator-fragment layout
  float c[4][8];
#pragma unroll
  for (int j = 0; j < 4; ++j)
#pragma unroll
    for (int v = 0; v < 8; ++v)
      c[j][v] = c0[(size_t)(row0 + v + half8) * HDIM + (j << 4) + m];

  __syncthreads();

  // ---- 2 LSTM steps: gates = [h|x|0] @ Wg_t, K=96 (3 k-steps) ----
  for (int t = 0; t < 2; ++t) {
    const __bf16* hin = (t == 0) ? actA : actB;
    const int inStride = (t == 0) ? 96 : 128;
    const v16bf a0 = load_a_frag(hin, inStride, 0,  lane);
    const v16bf a1 = load_a_frag(hin, inStride, 32, lane);
    const v16bf a2 = load_a_frag(hin, inStride, 64, lane);
#pragma unroll
    for (int j = 0; j < 4; ++j) {
      const float bi  = sm.bg[(j << 4) + m];
      const float bf_ = sm.bg[64  + (j << 4) + m];
      const float bgg = sm.bg[128 + (j << 4) + m];
      const float bo  = sm.bg[192 + (j << 4) + m];
      v8f ai, af, ag, ao;
#pragma unroll
      for (int v = 0; v < 8; ++v) { ai[v] = bi; af[v] = bf_; ag[v] = bgg; ao[v] = bo; }
      ai = wmma_b(a0, sm.Wg_t, 256, 0, j, lane, ai);
      ai = wmma_b(a1, sm.Wg_t, 256, 32, j, lane, ai);
      ai = wmma_b(a2, sm.Wg_t, 256, 64, j, lane, ai);
      af = wmma_b(a0, sm.Wg_t, 256, 0, j + 4, lane, af);
      af = wmma_b(a1, sm.Wg_t, 256, 32, j + 4, lane, af);
      af = wmma_b(a2, sm.Wg_t, 256, 64, j + 4, lane, af);
      ag = wmma_b(a0, sm.Wg_t, 256, 0, j + 8, lane, ag);
      ag = wmma_b(a1, sm.Wg_t, 256, 32, j + 8, lane, ag);
      ag = wmma_b(a2, sm.Wg_t, 256, 64, j + 8, lane, ag);
      ao = wmma_b(a0, sm.Wg_t, 256, 0, j + 12, lane, ao);
      ao = wmma_b(a1, sm.Wg_t, 256, 32, j + 12, lane, ao);
      ao = wmma_b(a2, sm.Wg_t, 256, 64, j + 12, lane, ao);
#pragma unroll
      for (int v = 0; v < 8; ++v) {
        const float iv = sigmoid_fast(ai[v]);
        const float fv = sigmoid_fast(af[v]);
        const float gv = tanh_fast(ag[v]);
        const float ov = sigmoid_fast(ao[v]);
        const float cv = fv * c[j][v] + iv * gv;
        c[j][v] = cv;
        const float hv = ov * tanh_fast(cv);
        if (t == 0) actB[(v + half8) * 128 + (j << 4) + m] = (__bf16)hv;             // h1 -> pong
        else        actA[(v + half8) * 96  + (j << 4) + m] = (__bf16)fmaxf(hv, 0.f); // relu(h2) -> ping
      }
#if __has_builtin(__builtin_amdgcn_sched_barrier)
      __builtin_amdgcn_sched_barrier(0);   // cap VGPR pressure across j-iterations
#endif
    }
  }

  // ---- MLP1 (64->256) split into two 128-col phases, MLP2 (256->64) partial
  //      sums held in registers so the pong buffer only needs 128 columns ----
  v8f acc2[4];
#pragma unroll
  for (int nt = 0; nt < 4; ++nt) {
    const float bn = sm.b2[(nt << 4) + m];
#pragma unroll
    for (int v = 0; v < 8; ++v) acc2[nt][v] = bn;
  }
  mlp_phase<2>(actA, 96, actB, 128, sm.W1_t, 256, sm.b1, 0, 8, lane);   // cols 0..127
  mlp2_accum(actB, sm.W2_t, 0, acc2, lane);
  mlp_phase<2>(actA, 96, actB, 128, sm.W1_t, 256, sm.b1, 8, 16, lane);  // cols 128..255
  mlp2_accum(actB, sm.W2_t, 128, acc2, lane);
#pragma unroll
  for (int nt = 0; nt < 4; ++nt)
#pragma unroll
    for (int v = 0; v < 8; ++v)
      actA[(v + half8) * 96 + (nt << 4) + m] = (__bf16)fmaxf(acc2[nt][v], 0.f);

  // ---- MLP3 (64->32), MLP4 (32->16) ----
  mlp_phase<2>(actA, 96, actB, 128, sm.W3_t, 32, sm.b3, 0, 2, lane);
  mlp_phase<1>(actB, 128, actA, 96, sm.W4_t, 16, sm.b4, 0, 1, lane);
  // zero cols 16..31 of actA so head K can be padded 16 -> 32
#pragma unroll
  for (int v = 0; v < 8; ++v) actA[(v + half8) * 96 + 16 + m] = (__bf16)0.f;

  // ---- head: single WMMA, N padded 3 -> 16 ----
  const float bhn = sm.bhp[m];
  v8f acc;
#pragma unroll
  for (int v = 0; v < 8; ++v) acc[v] = bhn;
  const v16bf ah = load_a_frag(actA, 96, 0, lane);
  acc = wmma_b(ah, sm.Wh_t, 16, 0, 0, lane, acc);
  if (m < 3) {
#pragma unroll
    for (int v = 0; v < 8; ++v)
      out[(size_t)(row0 + v + half8) * 3 + m] = acc[v];
  }
}

extern "C" void kernel_launch(void* const* d_in, const int* in_sizes, int n_in,
                              void* d_out, int out_size, void* d_ws, size_t ws_size,
                              hipStream_t stream) {
  const float* x    = (const float*)d_in[0];
  const float* h0   = (const float*)d_in[1];
  const float* c0   = (const float*)d_in[2];
  const float* W_ih = (const float*)d_in[3];
  const float* W_hh = (const float*)d_in[4];
  const float* b_ih = (const float*)d_in[5];
  const float* b_hh = (const float*)d_in[6];
  const float* W1   = (const float*)d_in[7];
  const float* b1   = (const float*)d_in[8];
  const float* W2   = (const float*)d_in[9];
  const float* b2   = (const float*)d_in[10];
  const float* W3   = (const float*)d_in[11];
  const float* b3   = (const float*)d_in[12];
  const float* W4   = (const float*)d_in[13];
  const float* b4   = (const float*)d_in[14];
  const float* Wh   = (const float*)d_in[15];
  const float* bh   = (const float*)d_in[16];
  float* out = (float*)d_out;

  const int B = in_sizes[1] / 64;          // h0 is [B, 64]
  const int blocks = B / (16 * WAVES);     // one 16-row tile per wave
  lstm_dqn_fused<<<blocks, TPB, 0, stream>>>(x, h0, c0, W_ih, W_hh, b_ih, b_hh,
                                             W1, b1, W2, b2, W3, b3, W4, b4,
                                             Wh, bh, out, B);
}
